// RefineModel_79096117723735
// MI455X (gfx1250) — compile-verified, matrix operand order checked by reading, not run on hardware
//
#include <hip/hip_runtime.h>
#include <hip/hip_bf16.h>

typedef __attribute__((ext_vector_type(16))) _Float16 v16h;
typedef __attribute__((ext_vector_type(8)))  _Float16 h8;
typedef __attribute__((ext_vector_type(8)))  float    v8f;

#define KK 10
#define NN 96
#define TT 40
#define HID 48

static __device__ __forceinline__ int iclampi(int v, int lo, int hi) {
    return v < lo ? lo : (v > hi ? hi : v);
}

// ---------------------------------------------------------------------------
// f16 WMMA GEMM with pre-swizzled B, software-pipelined (double-buffered
// fragments so chunk c+1's loads are in flight under chunk c's WMMA).
//   C[M x N] = A[M x KCH*32] * B + bias (optional relu)
// One wave (32 threads) per 16x16 C tile; KCH/NTILES compile-time.
// Bs layout: fragment order — for (chunk, tn): 32 lanes x 16 halves contiguous,
// so each lane's B fragment is one aligned 32-byte load.
// ---------------------------------------------------------------------------
template<int KCH, int NTILES, bool RELU>
__global__ void wmma_gemm(const _Float16* __restrict__ A, int lda,
                          const _Float16* __restrict__ Bs,
                          const float* __restrict__ bias,
                          float* __restrict__ C, int ldc)
{
    const int lane = threadIdx.x;            // blockDim.x == 32
    const int tile = blockIdx.x;
    const int tm = tile / NTILES;
    const int tn = tile % NTILES;

    const int rA = tm * 16 + (lane & 15);    // A row for this lane
    const int hi = (lane >> 4) * 8;          // K sub-offset: 0 or 8

    const _Float16* __restrict__ arow = A + (size_t)rA * lda;
    const v16h* __restrict__ bfrag =
        (const v16h*)(Bs + (size_t)tn * 512 + lane * 16);

    // A fragment: lane<16 holds K {kc+0..7, kc+16..23}; lane>=16 {+8..15,+24..31}
    auto loadA = [&](int c, v16h& a) {
        h8 lo = *(const h8*)(arow + c * 32 + hi);
        h8 hh = *(const h8*)(arow + c * 32 + 16 + hi);
        #pragma unroll
        for (int j = 0; j < 8; ++j) { a[j] = lo[j]; a[8 + j] = hh[j]; }
    };

    v16h a0, b0;
    v16h a1 = {}, b1 = {};
    loadA(0, a0);
    b0 = bfrag[0];

    v8f acc = {};
    #pragma unroll 6
    for (int c = 0; c < KCH; ++c) {
        if (c + 1 < KCH) {
            loadA(c + 1, a1);
            b1 = bfrag[(size_t)(c + 1) * NTILES * 32];
            if (KCH > 8)   // speculative: over-run prefetch is dropped safely
                __builtin_prefetch(arow + c * 32 + 512, 0, 1);
        }
        acc = __builtin_amdgcn_wmma_f32_16x16x32_f16(
                false, a0, false, b0, (short)0, acc, false, false);
        a0 = a1; b0 = b1;
    }

    // C/D layout: lane l -> col (l&15); rows (l>>4)*8 + 0..7 in acc[0..7]
    const int colC  = tn * 16 + (lane & 15);
    const float bv  = bias[colC];
    const int rbase = tm * 16 + (lane >> 4) * 8;
    #pragma unroll
    for (int r = 0; r < 8; ++r) {
        float v = acc[r] + bv;
        if (RELU) v = fmaxf(v, 0.0f);
        C[(size_t)(rbase + r) * ldc + colC] = v;
    }
}

// ---------------------------------------------------------------------------
// Weight prep: W is (Ncols, Korig) row-major f32 (a torch-style fc weight);
// emit f16 fragments for B = W^T (K x N), K zero-padded to KCH*32, in the
// swizzled order consumed by wmma_gemm: idx = ((chunk*NTILES + tn)*32 + lane)*16 + e
// ---------------------------------------------------------------------------
__global__ void prep_swz(const float* __restrict__ W, int Korig, int NTILES,
                         int total, _Float16* __restrict__ Bs)
{
    int idx = blockIdx.x * 256 + threadIdx.x;
    if (idx >= total) return;
    const int chunkstride = NTILES * 512;
    int chunk = idx / chunkstride;
    int r     = idx % chunkstride;
    int tn    = r / 512;
    int lane  = (r % 512) / 16;
    int e     = idx & 15;
    int hi    = (lane >> 4) * 8;
    int col   = tn * 16 + (lane & 15);
    int kk    = chunk * 32 + hi + (e < 8 ? e : 8 + e); // e>=8 -> +16+(e-8)
    float v = (kk < Korig) ? W[(size_t)col * Korig + kk] : 0.0f;
    Bs[idx] = (_Float16)v;
}

// ---------------------------------------------------------------------------
// conv1: (1,4,160,160) -> (16,80,80), k5 s2 p2, relu
// ---------------------------------------------------------------------------
__global__ void conv1_kernel(const float* __restrict__ img,
                             const float* __restrict__ w,
                             const float* __restrict__ b,
                             float* __restrict__ out)
{
    int idx = blockIdx.x * 256 + threadIdx.x;
    if (idx >= 16 * 80 * 80) return;
    int oc = idx / 6400, rem = idx % 6400;
    int oy = rem / 80, ox = rem % 80;
    float s = b[oc];
    for (int ic = 0; ic < 4; ++ic)
        for (int ky = 0; ky < 5; ++ky) {
            int iy = oy * 2 - 2 + ky;
            if (iy < 0 || iy >= 160) continue;
            for (int kx = 0; kx < 5; ++kx) {
                int ix = ox * 2 - 2 + kx;
                if (ix < 0 || ix >= 160) continue;
                s += img[((size_t)ic * 160 + iy) * 160 + ix] *
                     w[(((size_t)oc * 4 + ic) * 5 + ky) * 5 + kx];
            }
        }
    out[idx] = fmaxf(s, 0.0f);
}

// conv2: (16,80,80) -> (32,80,80), k5 s1 p2, relu
__global__ void conv2_kernel(const float* __restrict__ in,
                             const float* __restrict__ w,
                             const float* __restrict__ b,
                             float* __restrict__ out)
{
    int idx = blockIdx.x * 256 + threadIdx.x;
    if (idx >= 32 * 80 * 80) return;
    int oc = idx / 6400, rem = idx % 6400;
    int oy = rem / 80, ox = rem % 80;
    float s = b[oc];
    for (int ic = 0; ic < 16; ++ic)
        for (int ky = 0; ky < 5; ++ky) {
            int iy = oy - 2 + ky;
            if (iy < 0 || iy >= 80) continue;
            for (int kx = 0; kx < 5; ++kx) {
                int ix = ox - 2 + kx;
                if (ix < 0 || ix >= 80) continue;
                s += in[((size_t)ic * 80 + iy) * 80 + ix] *
                     w[(((size_t)oc * 16 + ic) * 5 + ky) * 5 + kx];
            }
        }
    out[idx] = fmaxf(s, 0.0f);
}

// velocity features: relu(vel @ fcvel_w.T + b) -> (K,T,N,16) f16
__global__ void fv_kernel(const float* __restrict__ y_path,
                          const float* __restrict__ cur,
                          const float* __restrict__ w,
                          const float* __restrict__ b,
                          _Float16* __restrict__ yfv)
{
    int idx = blockIdx.x * 256 + threadIdx.x;
    if (idx >= KK * TT * NN * 16) return;
    int o = idx & 15;
    int n = (idx >> 4) % NN;
    int t = (idx / (16 * NN)) % TT;
    int k = idx / (16 * NN * TT);
    const float* yp = y_path + (((size_t)k * TT + t) * NN + n) * 2;
    float px, py;
    if (t == 0) { px = cur[n * 2 + 0]; py = cur[n * 2 + 1]; }
    else {
        const float* pp = y_path + (((size_t)k * TT + t - 1) * NN + n) * 2;
        px = pp[0]; py = pp[1];
    }
    float vx = (yp[0] - px) * 10.0f;
    float vy = (yp[1] - py) * 10.0f;
    float v = fmaxf(vx * w[o * 2 + 0] + vy * w[o * 2 + 1] + b[o], 0.0f);
    yfv[idx] = (_Float16)v;
}

// h init: tile hx K times; f32 state + f16 padded (960x64) copy; zero score
__global__ void prep_h(const float* __restrict__ hx,
                       float* __restrict__ h, _Float16* __restrict__ h16,
                       float* __restrict__ score)
{
    int idx = blockIdx.x * 256 + threadIdx.x;
    if (idx >= 960 * 64) return;
    int row = idx / 64, col = idx % 64;
    if (col < 48) {
        float v = hx[(size_t)(row % NN) * 48 + col];
        h[(size_t)row * 48 + col] = v;
        h16[idx] = (_Float16)v;
    } else {
        h16[idx] = (_Float16)0.0f;
    }
    if (idx < 960) score[idx] = 0.0f;
}

// ---------------------------------------------------------------------------
// Spatial binning: one block per (k,i); produces sps row (1728) in f16.
// Reads h rows 0..95 (reference uses h[:N]) BEFORE this step's update.
// ---------------------------------------------------------------------------
__global__ void sps_kernel(const float* __restrict__ y_path, int t,
                           const float* __restrict__ h,
                           _Float16* __restrict__ sps)
{
    __shared__ int sbin[96];
    __shared__ int swin[36];
    __shared__ int scnt[36];
    const int kb = blockIdx.x;           // 0..959
    const int k = kb / NN, i = kb % NN;
    const int tid = threadIdx.x;         // 96 threads
    const float* locK = y_path + (((size_t)k * TT + t) * NN) * 2;
    const float xi = locK[i * 2 + 0], yi = locK[i * 2 + 1];
    if (tid < 96) {
        float cx = locK[tid * 2 + 0] - xi;
        float cy = locK[tid * 2 + 1] - yi;
        float dist = sqrtf(cx * cx + cy * cy);
        int bin = -1;
        if (tid != i && dist >= 0.5f && dist <= 4.0f) {
            float ds = fmaxf(dist, 1e-10f);
            float ct = fminf(fmaxf(cx / ds, -1.0f), 1.0f);
            float ac = acosf(ct);
            float th = (cy < 0.0f) ? (6.28318530718f - ac) : ac;
            int ub = iclampi((int)((dist - 0.5f) / 0.583333333f), 0, 5);
            int vb = iclampi((int)(th / 1.04719755120f), 0, 5);
            bin = ub * 6 + vb;
        }
        sbin[tid] = bin;
    }
    __syncthreads();
    if (tid < 36) {
        int cnt = 0, win = -1;
        for (int j = 0; j < 96; ++j)
            if (sbin[j] == tid) { cnt++; win = j; }
        scnt[tid] = cnt; swin[tid] = win;
    }
    __syncthreads();
    _Float16* out = sps + (size_t)kb * 1728;
    for (int idx = tid; idx < 1728; idx += 96) {
        int bin = idx / 48, d = idx % 48;
        int w = swin[bin];
        float v = 0.0f;
        if (w >= 0) {
            int c = scnt[bin]; if (c < 1) c = 1;
            v = h[(size_t)w * 48 + d] / (float)c;
        }
        out[idx] = (_Float16)v;
    }
}

// Assemble x_i row (96): [ feat(32 from fmap gather) | fv(16) | rhalf(48) ] -> f16
__global__ void xin_kernel(const float* __restrict__ y_path, int t,
                           const float* __restrict__ fmap,
                           const _Float16* __restrict__ yfv,
                           const float* __restrict__ rhalf,
                           _Float16* __restrict__ xi)
{
    int row = blockIdx.x;      // 960
    int c = threadIdx.x;       // 96
    int k = row / NN, n = row % NN;
    float v;
    if (c < 32) {
        const float* loc = y_path + (((size_t)k * TT + t) * NN + n) * 2;
        int ui = iclampi(40 - (int)loc[1], 0, 79);
        int vi = iclampi(40 - (int)loc[0], 0, 79);
        v = fmap[(size_t)c * 6400 + ui * 80 + vi];
    } else if (c < 48) {
        v = (float)yfv[(((size_t)k * TT + t) * NN + n) * 16 + (c - 32)];
    } else {
        v = rhalf[(size_t)row * 48 + (c - 48)];
    }
    xi[(size_t)row * 96 + c] = (_Float16)v;
}

// GRU gate fusion + per-step score accumulation. One block per row.
__global__ void gate_kernel(const float* __restrict__ gi,
                            const float* __restrict__ gh,
                            float* __restrict__ h, _Float16* __restrict__ h16,
                            const float* __restrict__ scw,
                            const float* __restrict__ scb,
                            float* __restrict__ score)
{
    __shared__ float sdot[48];
    int row = blockIdx.x;      // 960
    int d = threadIdx.x;       // 64 threads, 48 active
    if (d < 48) {
        const float* gir = gi + (size_t)row * 144;
        const float* ghr = gh + (size_t)row * 144;
        float ir = gir[d],      hr = ghr[d];
        float iz = gir[48 + d], hz = ghr[48 + d];
        float in = gir[96 + d], hn = ghr[96 + d];
        float r  = 1.0f / (1.0f + expf(-(ir + hr)));
        float z  = 1.0f / (1.0f + expf(-(iz + hz)));
        float ng = tanhf(in + r * hn);
        float hp = h[(size_t)row * 48 + d];
        float hnew = (1.0f - z) * ng + z * hp;
        h[(size_t)row * 48 + d] = hnew;
        h16[(size_t)row * 64 + d] = (_Float16)hnew;
        sdot[d] = hnew * scw[d];
    }
    __syncthreads();
    if (d == 0) {
        float s = scb[0];
        for (int j = 0; j < 48; ++j) s += sdot[j];
        score[row] += s;
    }
}

// delta_y = relu(h_last @ fcdy_w.T + b), scattered into (K,T,N,2) layout
__global__ void delta_kernel(const float* __restrict__ h,
                             const float* __restrict__ w,
                             const float* __restrict__ b,
                             float* __restrict__ out)
{
    int idx = blockIdx.x * 256 + threadIdx.x;
    if (idx >= 960 * 80) return;
    int row = idx / 80, o = idx % 80;
    const float* hr = h + (size_t)row * 48;
    const float* wr = w + (size_t)o * 48;
    float s = b[o];
    for (int j = 0; j < 48; ++j) s += hr[j] * wr[j];
    s = fmaxf(s, 0.0f);
    int c = o / TT, t = o % TT;      // fcdy output index o = c*T + t
    int k = row / NN, n = row % NN;
    out[(((size_t)k * TT + t) * NN + n) * 2 + c] = s;
}

__global__ void score_out(const float* __restrict__ score, float* __restrict__ out)
{
    int idx = blockIdx.x * 256 + threadIdx.x;
    if (idx < 960) out[idx] = score[idx];
}

// ---------------------------------------------------------------------------
extern "C" void kernel_launch(void* const* d_in, const int* in_sizes, int n_in,
                              void* d_out, int out_size, void* d_ws, size_t ws_size,
                              hipStream_t stream)
{
    (void)in_sizes; (void)n_in; (void)out_size; (void)ws_size;
    const float* hx   = (const float*)d_in[0];
    const float* cur  = (const float*)d_in[1];
    const float* ypth = (const float*)d_in[2];
    const float* img  = (const float*)d_in[3];
    const float* c1w  = (const float*)d_in[4];
    const float* c1b  = (const float*)d_in[5];
    const float* c2w  = (const float*)d_in[6];
    const float* c2b  = (const float*)d_in[7];
    const float* fvw  = (const float*)d_in[8];
    const float* fvb  = (const float*)d_in[9];
    const float* wih  = (const float*)d_in[10];
    const float* whh  = (const float*)d_in[11];
    const float* bih  = (const float*)d_in[12];
    const float* bhh  = (const float*)d_in[13];
    const float* scfw = (const float*)d_in[14];
    const float* scfb = (const float*)d_in[15];
    const float* scw  = (const float*)d_in[16];
    const float* scb  = (const float*)d_in[17];
    const float* dyw  = (const float*)d_in[18];
    const float* dyb  = (const float*)d_in[19];
    float* out = (float*)d_out;

    // bump-allocate workspace (256B aligned slices; ~7.8 MB total)
    char* base = (char*)d_ws;
    size_t off = 0;
    auto walloc = [&](size_t bytes) -> void* {
        void* p = base + off;
        off += (bytes + 255) & ~(size_t)255;
        return p;
    };
    float*    c1    = (float*)   walloc(102400u * 4);   // conv1 out (16,80,80)
    float*    fmap  = (float*)   walloc(204800u * 4);   // conv2 out (32,80,80)
    _Float16* yfv   = (_Float16*)walloc(614400u * 2);   // (K,T,N,16) f16
    _Float16* Bscf  = (_Float16*)walloc(82944u  * 2);   // swz (54ch x 3 tiles)
    _Float16* Bwih  = (_Float16*)walloc(13824u  * 2);   // swz (3ch x 9 tiles)
    _Float16* Bwhh  = (_Float16*)walloc(9216u   * 2);   // swz (2ch x 9 tiles)
    float*    h     = (float*)   walloc(46080u  * 4);   // (960,48) f32 state
    _Float16* h16   = (_Float16*)walloc(61440u  * 2);   // (960,64) f16 padded
    _Float16* sps   = (_Float16*)walloc(1658880u* 2);   // (960,1728)
    float*    rhalf = (float*)   walloc(46080u  * 4);   // (960,48)
    _Float16* xi    = (_Float16*)walloc(92160u  * 2);   // (960,96)
    float*    gi    = (float*)   walloc(138240u * 4);   // (960,144)
    float*    gh    = (float*)   walloc(138240u * 4);   // (960,144)
    float*    score = (float*)   walloc(960u    * 4);

    // one-time (per call) prep
    conv1_kernel<<<(102400 + 255) / 256, 256, 0, stream>>>(img, c1w, c1b, c1);
    conv2_kernel<<<(204800 + 255) / 256, 256, 0, stream>>>(c1, c2w, c2b, fmap);
    fv_kernel   <<<(614400 + 255) / 256, 256, 0, stream>>>(ypth, cur, fvw, fvb, yfv);
    prep_swz    <<<(82944  + 255) / 256, 256, 0, stream>>>(scfw, 1728, 3, 82944, Bscf);
    prep_swz    <<<(13824  + 255) / 256, 256, 0, stream>>>(wih,  96,   9, 13824, Bwih);
    prep_swz    <<<(9216   + 255) / 256, 256, 0, stream>>>(whh,  48,   9, 9216,  Bwhh);
    prep_h      <<<(61440  + 255) / 256, 256, 0, stream>>>(hx, h, h16, score);

    // sequential GRU chain: 6 kernels per step, GEMMs on WMMA
    for (int t = 0; t < TT; ++t) {
        sps_kernel<<<960, 96, 0, stream>>>(ypth, t, h, sps);
        // rhalf = relu(sps @ fcscf_w.T + b) : (960x1728)@(1728x48)
        wmma_gemm<54, 3, true><<<60 * 3, 32, 0, stream>>>(sps, 1728, Bscf,
                                                          scfb, rhalf, 48);
        xin_kernel<<<960, 96, 0, stream>>>(ypth, t, fmap, yfv, rhalf, xi);
        // gi = x_i @ wih.T + bih : (960x96)@(96x144)
        wmma_gemm<3, 9, false><<<60 * 9, 32, 0, stream>>>(xi, 96, Bwih,
                                                          bih, gi, 144);
        // gh = h @ whh.T + bhh : (960x64pad)@(64x144)
        wmma_gemm<2, 9, false><<<60 * 9, 32, 0, stream>>>(h16, 64, Bwhh,
                                                          bhh, gh, 144);
        gate_kernel<<<960, 64, 0, stream>>>(gi, gh, h, h16, scw, scb, score);
    }

    delta_kernel<<<(960 * 80 + 255) / 256, 256, 0, stream>>>(h, dyw, dyb, out);
    score_out   <<<(960 + 255) / 256, 256, 0, stream>>>(score, out + 76800);
}
